// CoreMemory_7610682048657
// MI455X (gfx1250) — compile-verified, matrix operand order checked by reading, not run on hardware
//
#include <hip/hip_runtime.h>

#define SS 4096
#define DIMD 1024
#define HEADS 16
#define HDIM 64

#define GSTR 40  // GEMM LDS slab row stride (u16): 80B, 16B-aligned, conflict-free
#define TSTR 72  // attention tile / P staging row stride (u16): 144B

typedef __attribute__((ext_vector_type(16))) __bf16 v16bf;
typedef __attribute__((ext_vector_type(8))) float v8f;
typedef __attribute__((ext_vector_type(4))) int v4i;
typedef unsigned short u16;
typedef unsigned int u32;

union FragU { v16bf v; uint4 q[2]; };

#if __has_builtin(__builtin_amdgcn_global_load_async_to_lds_b128)
#define HAVE_ASYNC_LDS 1
typedef __attribute__((address_space(1))) v4i* g_v4i_ptr;
typedef __attribute__((address_space(3))) v4i* l_v4i_ptr;
#else
#define HAVE_ASYNC_LDS 0
#endif

__device__ __forceinline__ void wait_asynccnt0() {
#if __has_builtin(__builtin_amdgcn_s_wait_asynccnt)
  __builtin_amdgcn_s_wait_asynccnt(0);
#else
  asm volatile("s_wait_asynccnt 0" ::: "memory");
#endif
}

__device__ __forceinline__ void copy16B(u16* l, const u16* g) {
#if HAVE_ASYNC_LDS
  __builtin_amdgcn_global_load_async_to_lds_b128(
      (g_v4i_ptr)(size_t)g, (l_v4i_ptr)(size_t)l, 0, 0);
#else
  *(uint4*)l = *(const uint4*)g;
#endif
}

__device__ __forceinline__ u16 f32_to_bf16(float f) {
  u32 u = __float_as_uint(f);
  u32 r = u + 0x7FFFu + ((u >> 16) & 1u);  // round-to-nearest-even
  return (u16)(r >> 16);
}

// Load a 16x32 bf16 operand fragment in the CDNA5 WMMA 16-bit A/B layout.
// base: row-major bf16, row stride `stride`; rows row0..row0+15, K-slab ks..ks+31.
// Lane L holds row (L&15); K values {kh*8..kh*8+7} then {16+kh*8..+7}, kh=L>>4.
__device__ __forceinline__ v16bf load_frag(const u16* base, int stride,
                                           int row0, int ks, int lane) {
  int m = lane & 15;
  int kh = lane >> 4;
  const u16* p = base + (size_t)(row0 + m) * stride + ks + kh * 8;
  FragU f;
  f.q[0] = *(const uint4*)(p);
  f.q[1] = *(const uint4*)(p + 16);
  return f.v;
}

__device__ __forceinline__ v8f wmma_bf16(v16bf a, v16bf b, v8f c) {
  return __builtin_amdgcn_wmma_f32_16x16x32_bf16(false, a, false, b, (short)0, c,
                                                 false, false);
}

// Stage ROWS x 32-u16 K-slab into LDS (row stride GSTR). 256 threads.
template <int ROWS>
__device__ __forceinline__ void stage_slab(u16* lds, const u16* g, int gstride) {
  int t = threadIdx.x;
#pragma unroll
  for (int i = 0; i < ROWS / 64; ++i) {
    int idx = i * 256 + t;
    int row = idx >> 2, seg = (idx & 3) << 3;
    copy16B(lds + row * GSTR + seg, g + (size_t)row * gstride + seg);
  }
}

// Stage 64 x 64-u16 tile into LDS (row stride TSTR). 256 threads.
__device__ __forceinline__ void stage_tile(u16* lds, const u16* g, int gstride) {
  int t = threadIdx.x;
#pragma unroll
  for (int i = 0; i < 2; ++i) {
    int idx = i * 256 + t;
    int row = idx >> 3, seg = (idx & 7) << 3;
    copy16B(lds + row * TSTR + seg, g + (size_t)row * gstride + seg);
  }
}

// ---------------- LayerNorm (f32 in -> bf16 out) ----------------
__global__ void __launch_bounds__(256)
ln_to_bf16(const float* __restrict__ x, const float* __restrict__ g,
           const float* __restrict__ b, u16* __restrict__ h) {
  __shared__ float red[256];
  int s = blockIdx.x, t = threadIdx.x;
  const float* row = x + (size_t)s * DIMD;
  float v[4];
  float sum = 0.f;
#pragma unroll
  for (int i = 0; i < 4; ++i) { v[i] = row[t + 256 * i]; sum += v[i]; }
  red[t] = sum;
  __syncthreads();
  for (int off = 128; off > 0; off >>= 1) {
    if (t < off) red[t] += red[t + off];
    __syncthreads();
  }
  float mean = red[0] * (1.f / DIMD);
  __syncthreads();
  float sq = 0.f;
#pragma unroll
  for (int i = 0; i < 4; ++i) { float d = v[i] - mean; sq += d * d; }
  red[t] = sq;
  __syncthreads();
  for (int off = 128; off > 0; off >>= 1) {
    if (t < off) red[t] += red[t + off];
    __syncthreads();
  }
  float inv = rsqrtf(red[0] * (1.f / DIMD) + 1e-5f);
  u16* hr = h + (size_t)s * DIMD;
#pragma unroll
  for (int i = 0; i < 4; ++i) {
    int d = t + 256 * i;
    hr[d] = f32_to_bf16((v[i] - mean) * inv * g[d] + b[d]);
  }
}

// ---------------- f32 -> bf16 weight conversion ----------------
__global__ void __launch_bounds__(256)
cvt_bf16(const float* __restrict__ src, u16* __restrict__ dst, int n) {
  int i = blockIdx.x * 256 + threadIdx.x;
  if (i < n) dst[i] = f32_to_bf16(src[i]);
}

// --------- LDS-staged WMMA GEMM: Y = A(MxK) * W(NxK)^T + bias ----------------
// Block: 8 waves = 128 rows x 256 cols; wave tile 64x64 (16 C tiles).
// Both operand slabs double-buffered in LDS via async-to-LDS; 24KB unique
// global traffic per K-slab per block (vs 64KB direct).
// FUSED=1: cols span [Q|K|V] (N=3072); Q,K -> [H][S][HD], V -> [H][HD][S].
// FUSED=0: single W, f32 row-major output.
template <int FUSED>
__global__ void __launch_bounds__(256)
gemm_staged(const u16* __restrict__ A,
            const u16* __restrict__ W0, const u16* __restrict__ W1,
            const u16* __restrict__ W2,
            const float* __restrict__ b0, const float* __restrict__ b1,
            const float* __restrict__ b2,
            u16* __restrict__ outQ, u16* __restrict__ outK,
            u16* __restrict__ outV, float* __restrict__ outF) {
  __shared__ u16 asl[2][128 * GSTR];
  __shared__ u16 bsl[2][256 * GSTR];

  int lane = threadIdx.x & 31, wave = threadIdx.x >> 5;
  int rowB = blockIdx.y * 128;
  int colG = blockIdx.x * 256;
  int which = FUSED ? (colG >> 10) : 0;  // uniform per block (256 | 1024)
  int colB = colG & 1023;
  const u16* W = (which == 0) ? W0 : ((which == 1) ? W1 : W2);
  const float* bias = (which == 0) ? b0 : ((which == 1) ? b1 : b2);
  const u16* Ab = A + (size_t)rowB * DIMD;
  const u16* Wb = W + (size_t)colB * DIMD;

  v8f acc[4][4];
#pragma unroll
  for (int i = 0; i < 4; ++i)
#pragma unroll
    for (int j = 0; j < 4; ++j) acc[i][j] = (v8f){};

  int r0 = (wave & 1) * 64;
  int c0 = (wave >> 1) * 64;

  stage_slab<128>(asl[0], Ab, DIMD);
  stage_slab<256>(bsl[0], Wb, DIMD);
#if HAVE_ASYNC_LDS
  wait_asynccnt0();
#endif
  __syncthreads();

  for (int ks = 0; ks < DIMD; ks += 32) {
    int cur = (ks >> 5) & 1;
    if (ks + 32 < DIMD) {  // stage next slab while computing on current
      stage_slab<128>(asl[cur ^ 1], Ab + ks + 32, DIMD);
      stage_slab<256>(bsl[cur ^ 1], Wb + ks + 32, DIMD);
    }
    v16bf a[4], b[4];
#pragma unroll
    for (int i = 0; i < 4; ++i) a[i] = load_frag(asl[cur], GSTR, r0 + i * 16, 0, lane);
#pragma unroll
    for (int j = 0; j < 4; ++j) b[j] = load_frag(bsl[cur], GSTR, c0 + j * 16, 0, lane);
#pragma unroll
    for (int i = 0; i < 4; ++i)
#pragma unroll
      for (int j = 0; j < 4; ++j) acc[i][j] = wmma_bf16(a[i], b[j], acc[i][j]);

    asm volatile("s_wait_dscnt 0" ::: "memory");
#if HAVE_ASYNC_LDS
    if (ks + 32 < DIMD) wait_asynccnt0();
#endif
    __syncthreads();
  }

  int nloc = lane & 15;
  int roff = (lane >> 4) << 3;
#pragma unroll
  for (int j = 0; j < 4; ++j) {
    int n = colB + c0 + j * 16 + nloc;
    float bval = bias[n];
    int hidx = n >> 6, hd = n & 63;
#pragma unroll
    for (int i = 0; i < 4; ++i) {
      int mb = rowB + r0 + i * 16 + roff;
#pragma unroll
      for (int r = 0; r < 8; ++r) {
        int mm = mb + r;
        float val = acc[i][j][r] + bval;
        if (FUSED) {
          u16 o = f32_to_bf16(val);
          if (which == 0) {
            outQ[((hidx * SS + mm) << 6) + hd] = o;
          } else if (which == 1) {
            outK[((hidx * SS + mm) << 6) + hd] = o;
          } else {
            outV[(hidx << 18) + (hd << 12) + mm] = o;  // [H][HD][S]
          }
        } else {
          outF[(size_t)mm * DIMD + n] = val;
        }
      }
    }
  }
}

// ---------------- Flash-style attention with block K/V LDS staging ----------
// Grid: (S/128, H). 8 waves per block, same head; 64-key tiles double-buffered
// via async-to-LDS; online softmax with deferred denominator reduction.
__global__ void __launch_bounds__(256)
attn_fwd(const u16* __restrict__ q, const u16* __restrict__ k,
         const u16* __restrict__ vt, const unsigned char* __restrict__ mask,
         u16* __restrict__ ctx) {
  __shared__ u16 kbuf[2][64 * TSTR];  // [key_local][hd]
  __shared__ u16 vbuf[2][64 * TSTR];  // [hd][key_local]
  __shared__ u16 pl[8 * 16 * TSTR];   // per-wave P staging

  int lane = threadIdx.x & 31;
  int wslot = threadIdx.x >> 5;
  int head = blockIdx.y;
  int q0 = blockIdx.x * 128 + wslot * 16;

  const u16* qh = q + ((size_t)head << 18);
  const u16* kh = k + ((size_t)head << 18);
  const u16* vh = vt + ((size_t)head << 18);
  u16* plw = pl + wslot * 16 * TSTR;

  v16bf qa0 = load_frag(qh, HDIM, q0, 0, lane);
  v16bf qa1 = load_frag(qh, HDIM, q0, 32, lane);

  v8f o0 = {}, o1 = {}, o2 = {}, o3 = {};
  float mrun[8], lrun[8];
#pragma unroll
  for (int r = 0; r < 8; ++r) { mrun[r] = -3.0e38f; lrun[r] = 0.f; }

  const float sc2 = 0.03125f * 1.4426950408889634f;  // 1/sqrt(1024) * log2(e)
  int nloc = lane & 15;
  int prow = (lane >> 4) << 3;
  int mA = lane & 15, khA = lane >> 4;

  stage_tile(kbuf[0], kh, HDIM);
  stage_tile(vbuf[0], vh, SS);
#if HAVE_ASYNC_LDS
  wait_asynccnt0();
#endif
  __syncthreads();

  for (int kt = 0; kt < SS; kt += 64) {
    int cur = (kt >> 6) & 1;
    if (kt + 64 < SS) {  // stage next tile while computing on current
      stage_tile(kbuf[cur ^ 1], kh + (size_t)(kt + 64) * HDIM, HDIM);
      stage_tile(vbuf[cur ^ 1], vh + (kt + 64), SS);
    }
    const u16* kb = kbuf[cur];
    const u16* vb = vbuf[cur];

    // scores: 16 queries x 64 keys -> four 16x16 C tiles (operands from LDS)
    v8f s[4];
#pragma unroll
    for (int t = 0; t < 4; ++t) {
      v16bf b0 = load_frag(kb, TSTR, t * 16, 0, lane);
      v16bf b1 = load_frag(kb, TSTR, t * 16, 32, lane);
      v8f st = {};
      st = wmma_bf16(qa0, b0, st);
      st = wmma_bf16(qa1, b1, st);
      s[t] = st;
    }
    bool ok[4];
#pragma unroll
    for (int t = 0; t < 4; ++t) ok[t] = mask[kt + t * 16 + nloc] != 0;

    float p[4][8];
#pragma unroll
    for (int r = 0; r < 8; ++r) {
      float v0 = ok[0] ? s[0][r] * sc2 : -3.0e38f;
      float v1 = ok[1] ? s[1][r] * sc2 : -3.0e38f;
      float v2 = ok[2] ? s[2][r] * sc2 : -3.0e38f;
      float v3 = ok[3] ? s[3][r] * sc2 : -3.0e38f;
      float tmax = fmaxf(fmaxf(v0, v1), fmaxf(v2, v3));
      tmax = fmaxf(tmax, __shfl_xor(tmax, 8, 32));
      tmax = fmaxf(tmax, __shfl_xor(tmax, 4, 32));
      tmax = fmaxf(tmax, __shfl_xor(tmax, 2, 32));
      tmax = fmaxf(tmax, __shfl_xor(tmax, 1, 32));
      float mnew = fmaxf(mrun[r], tmax);
      float c = exp2f(mrun[r] - mnew);
      p[0][r] = exp2f(v0 - mnew);
      p[1][r] = exp2f(v1 - mnew);
      p[2][r] = exp2f(v2 - mnew);
      p[3][r] = exp2f(v3 - mnew);
      // deferred denominator: per-lane partial, cross-lane reduced once at end
      lrun[r] = lrun[r] * c + (p[0][r] + p[1][r]) + (p[2][r] + p[3][r]);
      mrun[r] = mnew;
      o0[r] *= c; o1[r] *= c; o2[r] *= c; o3[r] *= c;
    }

    // P (C-layout f32, 16x64) -> LDS row-major bf16 -> A-fragments
#pragma unroll
    for (int r = 0; r < 8; ++r) {
#pragma unroll
      for (int t = 0; t < 4; ++t)
        plw[(prow + r) * TSTR + t * 16 + nloc] = f32_to_bf16(p[t][r]);
    }
    asm volatile("s_wait_dscnt 0" ::: "memory");  // wave-internal LDS exchange
    v16bf pfA, pfB;
    {
      FragU fa, fb;
      const u16* pr = plw + mA * TSTR + khA * 8;
      fa.q[0] = *(const uint4*)(pr);
      fa.q[1] = *(const uint4*)(pr + 16);
      fb.q[0] = *(const uint4*)(pr + 32);
      fb.q[1] = *(const uint4*)(pr + 48);
      pfA = fa.v;
      pfB = fb.v;
    }
    // ctx += P(16x64) * V(64x64), V operand from LDS ([hd][key] rows)
    o0 = wmma_bf16(pfA, load_frag(vb, TSTR, 0, 0, lane), o0);
    o1 = wmma_bf16(pfA, load_frag(vb, TSTR, 16, 0, lane), o1);
    o2 = wmma_bf16(pfA, load_frag(vb, TSTR, 32, 0, lane), o2);
    o3 = wmma_bf16(pfA, load_frag(vb, TSTR, 48, 0, lane), o3);
    o0 = wmma_bf16(pfB, load_frag(vb, TSTR, 0, 32, lane), o0);
    o1 = wmma_bf16(pfB, load_frag(vb, TSTR, 16, 32, lane), o1);
    o2 = wmma_bf16(pfB, load_frag(vb, TSTR, 32, 32, lane), o2);
    o3 = wmma_bf16(pfB, load_frag(vb, TSTR, 48, 32, lane), o3);

    asm volatile("s_wait_dscnt 0" ::: "memory");
#if HAVE_ASYNC_LDS
    if (kt + 64 < SS) wait_asynccnt0();
#endif
    __syncthreads();
  }

  float inv[8];
#pragma unroll
  for (int r = 0; r < 8; ++r) {
    float l = lrun[r];
    l += __shfl_xor(l, 8, 32);
    l += __shfl_xor(l, 4, 32);
    l += __shfl_xor(l, 2, 32);
    l += __shfl_xor(l, 1, 32);
    inv[r] = 1.f / l;
  }
  int rbase = q0 + prow;
#pragma unroll
  for (int r = 0; r < 8; ++r) {
    u16* crow = ctx + (size_t)(rbase + r) * DIMD + head * HDIM;
    crow[nloc] = f32_to_bf16(o0[r] * inv[r]);
    crow[16 + nloc] = f32_to_bf16(o1[r] * inv[r]);
    crow[32 + nloc] = f32_to_bf16(o2[r] * inv[r]);
    crow[48 + nloc] = f32_to_bf16(o3[r] * inv[r]);
  }
}

extern "C" void kernel_launch(void* const* d_in, const int* in_sizes, int n_in,
                              void* d_out, int out_size, void* d_ws, size_t ws_size,
                              hipStream_t stream) {
  (void)in_sizes; (void)n_in; (void)out_size; (void)ws_size;
  const float* x = (const float*)d_in[0];
  const unsigned char* mask = (const unsigned char*)d_in[1];
  const float* ln_g = (const float*)d_in[2];
  const float* ln_b = (const float*)d_in[3];
  const float* Wq = (const float*)d_in[4];
  const float* bq = (const float*)d_in[5];
  const float* Wk = (const float*)d_in[6];
  const float* bk = (const float*)d_in[7];
  const float* Wv = (const float*)d_in[8];
  const float* bv = (const float*)d_in[9];
  const float* Wo = (const float*)d_in[10];
  const float* bo = (const float*)d_in[11];
  float* out = (float*)d_out;

  const size_t MB = 1024 * 1024;
  char* ws = (char*)d_ws;
  u16* hbf  = (u16*)(ws + 0 * MB);   // [S][DIM] bf16
  u16* wqb  = (u16*)(ws + 8 * MB);   // [DIM][DIM] bf16
  u16* wkb  = (u16*)(ws + 10 * MB);
  u16* wvb  = (u16*)(ws + 12 * MB);
  u16* wob  = (u16*)(ws + 14 * MB);
  u16* qbf  = (u16*)(ws + 16 * MB);  // [H][S][HD]
  u16* kbf  = (u16*)(ws + 24 * MB);  // [H][S][HD]
  u16* vtb  = (u16*)(ws + 32 * MB);  // [H][HD][S]
  u16* ctxb = (u16*)(ws + 40 * MB);  // [S][DIM]

  ln_to_bf16<<<SS, 256, 0, stream>>>(x, ln_g, ln_b, hbf);

  int nW = DIMD * DIMD;
  cvt_bf16<<<nW / 256, 256, 0, stream>>>(Wq, wqb, nW);
  cvt_bf16<<<nW / 256, 256, 0, stream>>>(Wk, wkb, nW);
  cvt_bf16<<<nW / 256, 256, 0, stream>>>(Wv, wvb, nW);
  cvt_bf16<<<nW / 256, 256, 0, stream>>>(Wo, wob, nW);

  gemm_staged<1><<<dim3(12, 32), 256, 0, stream>>>(
      hbf, wqb, wkb, wvb, bq, bk, bv, qbf, kbf, vtb, nullptr);

  attn_fwd<<<dim3(SS / 128, HEADS), 256, 0, stream>>>(qbf, kbf, vtb, mask, ctxb);

  gemm_staged<0><<<dim3(4, 32), 256, 0, stream>>>(
      ctxb, wob, wob, wob, bo, bo, bo, nullptr, nullptr, nullptr, out);
}
